// RegGCN_45363444580421
// MI455X (gfx1250) — matrix operand order repaired and checked by reading, not compiled, not run on hardware
//
#include <hip/hip_runtime.h>
#include <hip/hip_bf16.h>

typedef __attribute__((ext_vector_type(16))) __bf16 v16bf;
typedef __attribute__((ext_vector_type(8)))  __bf16 v8bf;
typedef __attribute__((ext_vector_type(8)))  float  v8f;
typedef __attribute__((ext_vector_type(4)))  float  v4f;

#define NV   16384
#define DH   64
#define DOUTN 8

// ---------------------------------------------------------------------------
// helpers
// ---------------------------------------------------------------------------
static __device__ inline v16bf pack16(v4f f0, v4f f1, v4f f2, v4f f3) {
  v16bf r;
  r[0]  = (__bf16)f0.x; r[1]  = (__bf16)f0.y; r[2]  = (__bf16)f0.z; r[3]  = (__bf16)f0.w;
  r[4]  = (__bf16)f1.x; r[5]  = (__bf16)f1.y; r[6]  = (__bf16)f1.z; r[7]  = (__bf16)f1.w;
  r[8]  = (__bf16)f2.x; r[9]  = (__bf16)f2.y; r[10] = (__bf16)f2.z; r[11] = (__bf16)f2.w;
  r[12] = (__bf16)f3.x; r[13] = (__bf16)f3.y; r[14] = (__bf16)f3.z; r[15] = (__bf16)f3.w;
  return r;
}

static __device__ inline v8f wmma_bf16(v16bf a, v16bf b, v8f c) {
  // (neg_a, A, neg_b, B, c_mod, C, reuse_a, reuse_b)
  return __builtin_amdgcn_wmma_f32_16x16x32_bf16(false, a, false, b, (short)0, c,
                                                 false, false);
}

// A-fragment (16x32, bf16) from an fp32 row-major source row (regular loads).
// lane l<16 : row m0+l,  K in {c..c+15},  c = 16*(lane/16)
static __device__ inline v16bf load_a_frag(const float* __restrict__ row, int kbase) {
  v4f f0 = *(const v4f*)(row + kbase);
  v4f f1 = *(const v4f*)(row + kbase + 4);
  v4f f2 = *(const v4f*)(row + kbase + 8);
  v4f f3 = *(const v4f*)(row + kbase + 12);
  return pack16(f0, f1, f2, f3);
}

// Same, but non-temporal (TH=NT): used for the 1 GiB norm_adj stream which has
// no reuse and must not thrash L2 (which holds the reused qT/P operands).
static __device__ inline v16bf load_a_frag_nt(const float* __restrict__ row, int kbase) {
  v4f f0 = __builtin_nontemporal_load((const v4f*)(row + kbase));
  v4f f1 = __builtin_nontemporal_load((const v4f*)(row + kbase + 4));
  v4f f2 = __builtin_nontemporal_load((const v4f*)(row + kbase + 8));
  v4f f3 = __builtin_nontemporal_load((const v4f*)(row + kbase + 12));
  return pack16(f0, f1, f2, f3);
}

// ---------------------------------------------------------------------------
// K0: transpose weights to bf16 N-major (B-operand friendly) layouts
// ---------------------------------------------------------------------------
__global__ void prep_weights(const float* __restrict__ W1, const float* __restrict__ L1,
                             const float* __restrict__ W2, const float* __restrict__ L2,
                             const float* __restrict__ Wout,
                             __bf16* __restrict__ WT1, __bf16* __restrict__ LT1,
                             __bf16* __restrict__ WT2, __bf16* __restrict__ LT2,
                             __bf16* __restrict__ WoutT) {
  const int t = threadIdx.x;
  for (int i = t; i < DH * DH; i += 256) {
    const int k = i >> 6, n = i & 63;
    WT1[n * DH + k] = (__bf16)W1[k * DH + n];
    LT1[n * DH + k] = (__bf16)L1[k * DH + n];
    WT2[n * DH + k] = (__bf16)W2[k * DH + n];
    LT2[n * DH + k] = (__bf16)L2[k * DH + n];
  }
  // Wout is [8,64] == already N-major; pad to 16 rows with zeros.
  for (int i = t; i < 16 * DH; i += 256) {
    const int n = i >> 6, k = i & 63;
    WoutT[i] = (n < DOUTN) ? (__bf16)Wout[n * DH + k] : (__bf16)0.0f;
  }
}

// ---------------------------------------------------------------------------
// K1: P = x @ W (fp32 [NV,64]) ;  qT = (x @ L)^T (bf16 [64,NV])
// one wave per 16-row tile
// ---------------------------------------------------------------------------
__global__ void __launch_bounds__(256) proj_kernel(
    const float* __restrict__ x, const __bf16* __restrict__ WT,
    const __bf16* __restrict__ LT, float* __restrict__ P,
    __bf16* __restrict__ qT) {
  const int wave = (blockIdx.x * blockDim.x + threadIdx.x) >> 5;
  const int lane = threadIdx.x & 31;
  const int m0 = wave << 4;
  const int half = lane >> 4;
  const int l16 = lane & 15;
  const float* arow = x + (size_t)(m0 + l16) * DH;
  const int c0 = half << 4;

  v8f accW[4] = {};
  v8f accL[4] = {};
#pragma unroll
  for (int ks = 0; ks < 2; ++ks) {
    v16bf a = load_a_frag(arow, ks * 32 + c0);
#pragma unroll
    for (int t = 0; t < 4; ++t) {
      const int boff = (t * 16 + l16) * DH + ks * 32 + half * 16;
      v16bf bw = *(const v16bf*)(WT + boff);
      v16bf bl = *(const v16bf*)(LT + boff);
      accW[t] = wmma_bf16(a, bw, accW[t]);
      accL[t] = wmma_bf16(a, bl, accL[t]);
    }
  }
#pragma unroll
  for (int t = 0; t < 4; ++t) {
    const int n = t * 16 + l16;
    v8bf qb;
#pragma unroll
    for (int r = 0; r < 8; ++r) {
      const int m = m0 + r + half * 8;
      P[(size_t)m * DH + n] = accW[t][r];
      qb[r] = (__bf16)accL[t][r];
    }
    *(v8bf*)(qT + (size_t)n * NV + m0 + half * 8) = qb;  // 16B contiguous store
  }
}

// ---------------------------------------------------------------------------
// K2: xout = relu((adj @ q + P) * rw[region] + rb[region])    (the 1 GiB stream)
// one wave per 16-row tile, 512 k-steps of 16x16x32 bf16 WMMA on 4 n-tiles
// ---------------------------------------------------------------------------
__global__ void __launch_bounds__(256) adj_kernel(
    const float* __restrict__ adj, const __bf16* __restrict__ qT,
    const float* __restrict__ P, const int* __restrict__ region,
    const float* __restrict__ rw, const float* __restrict__ rb,
    float* __restrict__ xout) {
  const int wave = (blockIdx.x * blockDim.x + threadIdx.x) >> 5;
  const int lane = threadIdx.x & 31;
  const int m0 = wave << 4;
  const int half = lane >> 4;
  const int l16 = lane & 15;
  const float* arow = adj + (size_t)(m0 + l16) * NV + (half << 4);
  const __bf16* b0 = qT + (size_t)l16 * NV + (half << 4);

  v8f acc[4] = {};
#pragma unroll 2
  for (int k0 = 0; k0 < NV; k0 += 32) {
    __builtin_prefetch(arow + k0 + 1024, 0, 0);  // stream A ahead
    v16bf a = load_a_frag_nt(arow, k0);          // NT: no reuse, keep L2 clean
#pragma unroll
    for (int t = 0; t < 4; ++t) {
      v16bf b = *(const v16bf*)(b0 + (size_t)t * 16 * NV + k0);
      acc[t] = wmma_bf16(a, b, acc[t]);
    }
  }

  int regs[8];
#pragma unroll
  for (int r = 0; r < 8; ++r) regs[r] = region[m0 + r + half * 8];
#pragma unroll
  for (int t = 0; t < 4; ++t) {
    const int n = t * 16 + l16;
#pragma unroll
    for (int r = 0; r < 8; ++r) {
      const int m = m0 + r + half * 8;
      const float z = acc[t][r] + P[(size_t)m * DH + n];
      const float v = z * rw[regs[r] * DH + n] + rb[regs[r] * DH + n];
      xout[(size_t)m * DH + n] = v > 0.0f ? v : 0.0f;
    }
  }
}

// ---------------------------------------------------------------------------
// K3: out = x @ Wout^T + bout   (N padded 8 -> 16, only n<8 stored)
// ---------------------------------------------------------------------------
__global__ void __launch_bounds__(256) out_kernel(
    const float* __restrict__ x, const __bf16* __restrict__ WoutT,
    const float* __restrict__ bout, float* __restrict__ out) {
  const int wave = (blockIdx.x * blockDim.x + threadIdx.x) >> 5;
  const int lane = threadIdx.x & 31;
  const int m0 = wave << 4;
  const int half = lane >> 4;
  const int l16 = lane & 15;
  const float* arow = x + (size_t)(m0 + l16) * DH;
  const int c0 = half << 4;

  v8f acc = {};
#pragma unroll
  for (int ks = 0; ks < 2; ++ks) {
    v16bf a = load_a_frag(arow, ks * 32 + c0);
    v16bf b = *(const v16bf*)(WoutT + l16 * DH + ks * 32 + half * 16);
    acc = wmma_bf16(a, b, acc);
  }
  if (l16 < DOUTN) {
    const float bb = bout[l16];
#pragma unroll
    for (int r = 0; r < 8; ++r) {
      const int m = m0 + r + half * 8;
      out[(size_t)m * DOUTN + l16] = acc[r] + bb;
    }
  }
}

// ---------------------------------------------------------------------------
extern "C" void kernel_launch(void* const* d_in, const int* in_sizes, int n_in,
                              void* d_out, int out_size, void* d_ws, size_t ws_size,
                              hipStream_t stream) {
  const float* h    = (const float*)d_in[0];
  const float* adj  = (const float*)d_in[1];
  const int*   reg  = (const int*)  d_in[2];
  const float* W1   = (const float*)d_in[3];
  const float* L1   = (const float*)d_in[4];
  const float* rw1  = (const float*)d_in[5];
  const float* rb1  = (const float*)d_in[6];
  const float* W2   = (const float*)d_in[7];
  const float* L2   = (const float*)d_in[8];
  const float* rw2  = (const float*)d_in[9];
  const float* rb2  = (const float*)d_in[10];
  const float* Wout = (const float*)d_in[11];
  const float* bout = (const float*)d_in[12];
  float* out = (float*)d_out;

  char* ws = (char*)d_ws;
  __bf16* WT1   = (__bf16*)(ws + 0);
  __bf16* LT1   = (__bf16*)(ws + 8192);
  __bf16* WT2   = (__bf16*)(ws + 16384);
  __bf16* LT2   = (__bf16*)(ws + 24576);
  __bf16* WoutT = (__bf16*)(ws + 32768);
  float*  P     = (float*) (ws + 65536);                         // 4 MB
  __bf16* qT    = (__bf16*)(ws + 65536 + 4u * 1024 * 1024);      // 2 MB
  float*  x1    = (float*) (ws + 65536 + 6u * 1024 * 1024);      // 4 MB
  float*  x2    = (float*) (ws + 65536 + 10u * 1024 * 1024);     // 4 MB

  const dim3 blk(256);
  const dim3 grd(NV / 16 / 8);  // 1024 wave-tiles / 8 waves per block = 128

  prep_weights<<<1, 256, 0, stream>>>(W1, L1, W2, L2, Wout, WT1, LT1, WT2, LT2, WoutT);

  // layer 1
  proj_kernel<<<grd, blk, 0, stream>>>(h, WT1, LT1, P, qT);
  adj_kernel <<<grd, blk, 0, stream>>>(adj, qT, P, reg, rw1, rb1, x1);
  // layer 2
  proj_kernel<<<grd, blk, 0, stream>>>(x1, WT2, LT2, P, qT);
  adj_kernel <<<grd, blk, 0, stream>>>(adj, qT, P, reg, rw2, rb2, x2);
  // output head
  out_kernel <<<grd, blk, 0, stream>>>(x2, WoutT, bout, out);
}
